// JKNet_GATConv_23089744183639
// MI455X (gfx1250) — matrix operand order repaired and checked by reading, not compiled, not run on hardware
//
#include <hip/hip_runtime.h>
#include <hip/hip_bf16.h>

typedef __attribute__((ext_vector_type(16))) _Float16 v16h;
typedef __attribute__((ext_vector_type(8)))  float    v8f;

#define GAT_HEADS 4
#define GAT_HID   64
#define GAT_WIDTH 256
#define NEG_SLOPE 0.2f

// ---- monotone float<->int mapping for atomicMax on signed ints ------------
__device__ __forceinline__ int fmap(float f) {
  int i = __float_as_int(f);
  return i >= 0 ? i : (i ^ 0x7fffffff);
}
__device__ __forceinline__ float funmap(int i) {
  int j = i >= 0 ? i : (i ^ 0x7fffffff);
  return __int_as_float(j);
}
#define NEG_INF_MAP 0x807FFFFF   // fmap(-inf)

__device__ __forceinline__ float leaky(float x) { return x > 0.f ? x : NEG_SLOPE * x; }

// ---- weight convert (f32 -> f16, optional column pad with zeros) ----------
__global__ void cvt_pad_f16(const float* __restrict__ in, _Float16* __restrict__ out,
                            int rows, int cin, int cout) {
  int idx = blockIdx.x * blockDim.x + threadIdx.x;
  if (idx >= rows * cout) return;
  int r = idx / cout, c = idx - r * cout;
  out[idx] = (c < cin) ? (_Float16)in[r * cin + c] : (_Float16)0.f;
}

// ---- WMMA GEMM: C[M x Nt] = A[M x K] * B[K x Nt] (+bias), Nt = gridDim.y*64
// A: f32 or f16 row-major (lda), B: f16 row-major (ldb), C: f32 (ldc).
template <typename TA>
__global__ void gemm_wmma(const TA* __restrict__ A, int lda,
                          const _Float16* __restrict__ B, int ldb,
                          float* __restrict__ C, int ldc,
                          int M, int K,
                          const float* __restrict__ bias, int nstore) {
  __shared__ _Float16 As[64][34];   // 64 rows x 32 k (padded)
  __shared__ _Float16 Bs[32][66];   // 32 k x 64 n (padded)

  const int tid  = threadIdx.x;        // 0..127, 4 waves
  const int wave = tid >> 5;
  const int lane = tid & 31;
  const int half = lane >> 4;
  const int r16  = lane & 15;
  const int m0 = blockIdx.x * 64;
  const int n0 = blockIdx.y * 64;

  v8f acc[4];
  #pragma unroll
  for (int i = 0; i < 4; i++)
    #pragma unroll
    for (int j = 0; j < 8; j++) acc[i][j] = 0.f;

  const int arow = tid >> 1;           // 0..63
  const int acol = (tid & 1) * 16;
  const int brow = tid >> 2;           // 0..31
  const int bcol = (tid & 3) * 16;

  for (int k0 = 0; k0 < K; k0 += 32) {
    // stage A tile (convert to f16 on the fly)
    {
      int gr = m0 + arow;
      if (gr < M) {
        const TA* ap = A + (long)gr * lda + k0 + acol;
        #pragma unroll
        for (int i = 0; i < 16; i++) As[arow][acol + i] = (_Float16)(float)ap[i];
      } else {
        #pragma unroll
        for (int i = 0; i < 16; i++) As[arow][acol + i] = (_Float16)0.f;
      }
    }
    // stage B tile
    {
      const _Float16* bp = B + (long)(k0 + brow) * ldb + n0 + bcol;
      #pragma unroll
      for (int i = 0; i < 16; i++) Bs[brow][bcol + i] = bp[i];
    }
    __syncthreads();

    // A fragment: lane<16 rows M=r16; element e -> k = (e&7) + (e>>3)*16 + half*8
    v16h af;
    #pragma unroll
    for (int e = 0; e < 16; e++) {
      int kk = (e & 7) + ((e >> 3) << 4) + half * 8;
      af[e] = As[wave * 16 + r16][kk];
    }
    #pragma unroll
    for (int nt = 0; nt < 4; nt++) {
      // B fragment: N = r16; element e -> k = (e&7) + (e>>3)*16 + half*8
      v16h bf;
      #pragma unroll
      for (int e = 0; e < 16; e++) bf[e] = Bs[(e & 7) + ((e >> 3) << 4) + half * 8][nt * 16 + r16];
      acc[nt] = __builtin_amdgcn_wmma_f32_16x16x32_f16(
          false, af, false, bf, (short)0, acc[nt], false, false);
    }
    __syncthreads();
  }

  // D layout: N = r16, M = rr + half*8 within the wave's 16-row slice
  #pragma unroll
  for (int nt = 0; nt < 4; nt++) {
    int col = n0 + nt * 16 + r16;
    if (col >= nstore) continue;
    float bv = bias ? bias[col] : 0.f;
    #pragma unroll
    for (int rr = 0; rr < 8; rr++) {
      int row = m0 + wave * 16 + half * 8 + rr;
      if (row < M) C[(long)row * ldc + col] = acc[nt][rr] + bv;
    }
  }
}

// ---- per-node attention logits: es/ed[n][h] = sum_c H[n,h,c]*a[h,c] -------
__global__ void node_logits(const float* __restrict__ Hg,
                            const float* __restrict__ as_, const float* __restrict__ ad_,
                            float* __restrict__ es, float* __restrict__ ed) {
  int node = blockIdx.x;
  int h = threadIdx.x >> 5;       // wave = head
  int l = threadIdx.x & 31;
  const float* hp = Hg + (long)node * GAT_WIDTH + h * GAT_HID;
  float2 hv = *(const float2*)(hp + 2 * l);
  float2 av = *(const float2*)(as_ + h * GAT_HID + 2 * l);
  float2 dv = *(const float2*)(ad_ + h * GAT_HID + 2 * l);
  float s = hv.x * av.x + hv.y * av.y;
  float d = hv.x * dv.x + hv.y * dv.y;
  #pragma unroll
  for (int off = 16; off > 0; off >>= 1) {
    s += __shfl_xor(s, off, 32);
    d += __shfl_xor(d, off, 32);
  }
  if (l == 0) {
    es[node * GAT_HEADS + h] = s;
    ed[node * GAT_HEADS + h] = d;
  }
}

// ================= CSR build (graph is fixed across layers) ================
__global__ void zero_i32(int* __restrict__ p, int n) {
  int i = blockIdx.x * blockDim.x + threadIdx.x;
  if (i < n) p[i] = 0;
}

__global__ void count_deg(const int* __restrict__ ei, int E, int Etot,
                          int* __restrict__ deg) {
  int e = blockIdx.x * blockDim.x + threadIdx.x;
  if (e >= Etot) return;
  int d = (e < E) ? ei[E + e] : (e - E);
  atomicAdd(&deg[d], 1);
}

// single-block exclusive scan -> rowptr[0..N], cursor initialized to rowptr
__global__ void scan_rowptr(const int* __restrict__ deg, int* __restrict__ rowptr,
                            int* __restrict__ cursor, int N) {
  __shared__ int buf[256];
  __shared__ int carry;
  if (threadIdx.x == 0) carry = 0;
  __syncthreads();
  for (int base = 0; base < N; base += 256) {
    int i = base + (int)threadIdx.x;
    int v = (i < N) ? deg[i] : 0;
    buf[threadIdx.x] = v;
    __syncthreads();
    #pragma unroll
    for (int off = 1; off < 256; off <<= 1) {
      int t = (threadIdx.x >= (unsigned)off) ? buf[threadIdx.x - off] : 0;
      __syncthreads();
      buf[threadIdx.x] += t;
      __syncthreads();
    }
    int excl = buf[threadIdx.x] - v;
    if (i < N) { rowptr[i] = carry + excl; cursor[i] = carry + excl; }
    __syncthreads();
    if (threadIdx.x == 255) carry += buf[255];
    __syncthreads();
  }
  if (threadIdx.x == 0) rowptr[N] = carry;
}

// scatter src ids into dst-sorted order
__global__ void scatter_edges(const int* __restrict__ ei, int E, int Etot,
                              int* __restrict__ cursor, int* __restrict__ srcs) {
  int e = blockIdx.x * blockDim.x + threadIdx.x;
  if (e >= Etot) return;
  int s, d;
  if (e < E) { s = ei[e]; d = ei[E + e]; }
  else       { s = d = e - E; }
  int pos = atomicAdd(&cursor[d], 1);
  srcs[pos] = s;
}

// ==== fused per-node softmax + weighted gather (no global atomics) =========
__global__ void gat_aggregate(const int* __restrict__ rowptr, const int* __restrict__ srcs,
                              const float* __restrict__ es, const float* __restrict__ ed,
                              const float* __restrict__ Hg, float* __restrict__ out) {
  __shared__ int   miS[GAT_HEADS];   // running max (mapped int)
  __shared__ float dS[GAT_HEADS];    // running exp-sum
  const int node = blockIdx.x;
  const int t = threadIdx.x;         // 0..255
  const int beg = rowptr[node], end = rowptr[node + 1];

  if (t < GAT_HEADS) { miS[t] = NEG_INF_MAP; dS[t] = 0.f; }
  __syncthreads();

  // phase A: per-head max then exp-sum over this node's edge list
  {
    int h = t & 3, slot = t >> 2;    // 64 slots per head
    float edv = ed[node * GAT_HEADS + h];
    float lmax = -3.0e38f;
    for (int j = beg + slot; j < end; j += 64)
      lmax = fmaxf(lmax, leaky(es[srcs[j] * GAT_HEADS + h] + edv));
    atomicMax(&miS[h], fmap(lmax));
    __syncthreads();
    float m = funmap(miS[h]);
    float lsum = 0.f;
    for (int j = beg + slot; j < end; j += 64)
      lsum += __expf(leaky(es[srcs[j] * GAT_HEADS + h] + edv) - m);
    atomicAdd(&dS[h], lsum);
  }
  __syncthreads();

  // phase B: weighted coalesced gather; one output channel per thread
  {
    int h = t >> 6;
    float edv = ed[node * GAT_HEADS + h];
    float m   = funmap(miS[h]);
    float den = dS[h];
    float acc = 0.f;
    for (int j = beg; j < end; j++) {
      int s = srcs[j];
      float alpha = __expf(leaky(es[s * GAT_HEADS + h] + edv) - m) / den;
      acc += Hg[(long)s * GAT_WIDTH + t] * alpha;
    }
    out[(long)node * GAT_WIDTH + t] = acc;
  }
}

// ---- finalize: h = relu(h + bias); keep f32 for next layer + f16 concat ---
__global__ void finalize(float* __restrict__ Hin, const float* __restrict__ b,
                         _Float16* __restrict__ hcat, int layer, int Ntot) {
  int idx = blockIdx.x * blockDim.x + threadIdx.x;
  if (idx >= Ntot * GAT_WIDTH) return;
  int c = idx & (GAT_WIDTH - 1);
  int row = idx >> 8;
  float v = Hin[idx] + b[c];
  v = fmaxf(v, 0.f);
  Hin[idx] = v;
  hcat[(long)row * (3 * GAT_WIDTH) + layer * GAT_WIDTH + c] = (_Float16)v;
}

extern "C" void kernel_launch(void* const* d_in, const int* in_sizes, int n_in,
                              void* d_out, int out_size, void* d_ws, size_t ws_size,
                              hipStream_t stream) {
  const float* x   = (const float*)d_in[0];
  const int*   ei  = (const int*)d_in[1];
  const float* W1  = (const float*)d_in[2];
  const float* as1 = (const float*)d_in[3];
  const float* ad1 = (const float*)d_in[4];
  const float* b1  = (const float*)d_in[5];
  const float* W2  = (const float*)d_in[6];
  const float* as2 = (const float*)d_in[7];
  const float* ad2 = (const float*)d_in[8];
  const float* b2  = (const float*)d_in[9];
  const float* W3  = (const float*)d_in[10];
  const float* as3 = (const float*)d_in[11];
  const float* ad3 = (const float*)d_in[12];
  const float* b3  = (const float*)d_in[13];
  const float* Wo  = (const float*)d_in[14];
  const float* bo  = (const float*)d_in[15];

  const int N    = in_sizes[0] / 512;   // nodes
  const int E    = in_sizes[1] / 2;     // raw edges
  const int Etot = E + N;               // + self loops

  // ---- carve workspace ----
  char* p = (char*)d_ws;
  auto carve = [&](size_t bytes) -> char* {
    char* r = p; p += (bytes + 255) & ~(size_t)255; return r;
  };
  _Float16* w1h    = (_Float16*)carve((size_t)512 * 256 * 2);
  _Float16* w2h    = (_Float16*)carve((size_t)256 * 256 * 2);
  _Float16* w3h    = (_Float16*)carve((size_t)256 * 256 * 2);
  _Float16* woh    = (_Float16*)carve((size_t)768 * 64 * 2);
  float*    Hg     = (float*)carve((size_t)N * 256 * 4);   // per-layer x@W
  float*    Hin    = (float*)carve((size_t)N * 256 * 4);   // aggregated / next input
  float*    es     = (float*)carve((size_t)N * 4 * 4);
  float*    ed     = (float*)carve((size_t)N * 4 * 4);
  int*      deg    = (int*)carve((size_t)N * 4);
  int*      cursor = (int*)carve((size_t)N * 4);
  int*      rowptr = (int*)carve((size_t)(N + 1) * 4);
  int*      srcs   = (int*)carve((size_t)Etot * 4);
  _Float16* hcat   = (_Float16*)carve((size_t)N * 768 * 2);
  (void)ws_size; (void)n_in; (void)out_size;

  // ---- weights to f16 (Wo padded 40 -> 64 cols) ----
  cvt_pad_f16<<<(512 * 256 + 255) / 256, 256, 0, stream>>>(W1, w1h, 512, 256, 256);
  cvt_pad_f16<<<(256 * 256 + 255) / 256, 256, 0, stream>>>(W2, w2h, 256, 256, 256);
  cvt_pad_f16<<<(256 * 256 + 255) / 256, 256, 0, stream>>>(W3, w3h, 256, 256, 256);
  cvt_pad_f16<<<(768 * 64 + 255) / 256, 256, 0, stream>>>(Wo, woh, 768, 40, 64);

  // ---- build dst-sorted CSR once (shared by all 3 layers) ----
  zero_i32<<<(N + 255) / 256, 256, 0, stream>>>(deg, N);
  count_deg<<<(Etot + 255) / 256, 256, 0, stream>>>(ei, E, Etot, deg);
  scan_rowptr<<<1, 256, 0, stream>>>(deg, rowptr, cursor, N);
  scatter_edges<<<(Etot + 255) / 256, 256, 0, stream>>>(ei, E, Etot, cursor, srcs);

  const float* Asrc[3] = {as1, as2, as3};
  const float* Adst[3] = {ad1, ad2, ad3};
  const float* Bias[3] = {b1, b2, b3};
  _Float16*    Wh[3]   = {w1h, w2h, w3h};

  const int nw = N * GAT_WIDTH;

  for (int l = 0; l < 3; l++) {
    const float* Ain = (l == 0) ? x : Hin;
    int K = (l == 0) ? 512 : 256;
    dim3 grid((N + 63) / 64, 4);
    gemm_wmma<float><<<grid, 128, 0, stream>>>(Ain, K, Wh[l], 256, Hg, 256,
                                               N, K, nullptr, 256);
    node_logits<<<N, 128, 0, stream>>>(Hg, Asrc[l], Adst[l], es, ed);
    gat_aggregate<<<N, 256, 0, stream>>>(rowptr, srcs, es, ed, Hg, Hin);
    finalize<<<(nw + 255) / 256, 256, 0, stream>>>(Hin, Bias[l], hcat, l, N);
  }

  // ---- JK concat projection: [N,768] @ [768,40] + bo ----
  dim3 go((N + 63) / 64, 1);
  gemm_wmma<_Float16><<<go, 128, 0, stream>>>(hcat, 768, woh, 64,
                                              (float*)d_out, 40, N, 768, bo, 40);
}
